// GNNWithEdgeFeatures_63144609186302
// MI455X (gfx1250) — matrix operand order repaired and checked by reading, not compiled, not run on hardware
//
#include <hip/hip_runtime.h>
#include <hip/hip_bf16.h>

typedef __attribute__((ext_vector_type(16))) _Float16 v16h;
typedef __attribute__((ext_vector_type(8)))  _Float16 v8h;
typedef __attribute__((ext_vector_type(8)))  float    v8f;

#define F_IN 16
#define F_E  8
#define HDIM 64
#define KDIM 1024   // H * F_IN
#define HEADS 4
#define DHEAD 16
#define NGRP 64
#define EPSLN 1e-5f

static __device__ inline v8f v8f_zero() {
    v8f z;
#pragma unroll
    for (int i = 0; i < 8; ++i) z[i] = 0.f;
    return z;
}

static __device__ inline void atomicMaxFloat(float* addr, float val) {
    int iv = __float_as_int(val);
    if (iv >= 0) atomicMax((int*)addr, iv);
    else atomicMin((unsigned int*)addr, (unsigned int)iv);
}

// ---------------- prep: convert W2 -> f16 transposed [n][k], gat_w -> f16 [n][k]
__global__ void k_prep(const float* w2, const float* gatw, _Float16* w2t, _Float16* gwt) {
    int i = blockIdx.x * 256 + threadIdx.x;
    const int T1 = KDIM * KDIM;
    if (i < T1) {
        int k = i >> 10, n = i & 1023;            // w2[k][n]
        w2t[(size_t)n * KDIM + k] = (_Float16)w2[i];
    } else if (i < T1 + HDIM * HDIM) {
        int j = i - T1;
        int k = j >> 6, n = j & 63;               // gatw[k][n]
        gwt[n * HDIM + k] = (_Float16)gatw[j];
    }
}

// ---------------- workspace init
__global__ void k_zero(float* p, long n) {
    long i = (long)blockIdx.x * 256 + threadIdx.x;
    long stride = (long)gridDim.x * 256;
    for (; i < n; i += stride) p[i] = 0.f;
}
__global__ void k_fill(float* p, long n, float v) {
    long i = (long)blockIdx.x * 256 + threadIdx.x;
    if (i < n) p[i] = v;
}

// ---------------- fused NNConv: edge MLP -> WMMA GEMM -> per-edge contraction -> scatter
// 512 threads = 16 waves; wave w owns output columns [w*64, w*64+64) (4 N-tiles, f == w)
__global__ __launch_bounds__(512) void k_nnconv(
    const float* __restrict__ x, const int* __restrict__ ei,
    const float* __restrict__ ea, const float* __restrict__ w1,
    const float* __restrict__ b1, const float* __restrict__ b2,
    const _Float16* __restrict__ w2t,
    float* __restrict__ agg, float* __restrict__ deg, int Nn, int Ee) {

    __shared__ float sh_ea[32][F_E];
    __shared__ float sh_x[32][F_IN];
    __shared__ __align__(16) _Float16 sh_hid[32][128];
    __shared__ float sh_msg[32 * HDIM];

    const int tid  = threadIdx.x;
    const int lane = tid & 31;
    const int w    = tid >> 5;          // wave 0..15
    const int half = lane >> 4;         // 0/1
    const int l15  = lane & 15;
    const long eb  = (long)blockIdx.x * 32;
    const int* srcI = ei;
    const int* dstI = ei + Ee;

    // stage edge_attr [32][8] and x[src] [32][16]
    if (tid < 32 * F_E) {
        int e = tid >> 3, f = tid & 7;
        long eg = eb + e;
        sh_ea[e][f] = (eg < Ee) ? ea[eg * F_E + f] : 0.f;
    }
    {
        int i = tid;                              // 512 == 32*16
        int e = i >> 4, f = i & 15;
        long eg = eb + e;
        float v = 0.f;
        if (eg < Ee) { int s = srcI[eg]; v = x[(long)s * F_IN + f]; }
        sh_x[e][f] = v;
    }
    for (int i = tid; i < 32 * HDIM; i += 512) sh_msg[i] = 0.f;
    __syncthreads();

    v8f acc[2][4];
#pragma unroll
    for (int m = 0; m < 2; ++m)
#pragma unroll
        for (int t = 0; t < 4; ++t) acc[m][t] = v8f_zero();

    auto loadB = [&](int c, int s, int t) -> v16h {
        const _Float16* bp = w2t + (size_t)(w * 64 + t * 16 + l15) * KDIM
                                 + (c * 128 + s * 32 + half * 8);
        v8h blo = *(const v8h*)bp;
        v8h bhi = *(const v8h*)(bp + 16);
        return __builtin_shufflevector(blo, bhi, 0,1,2,3,4,5,6,7,8,9,10,11,12,13,14,15);
    };

    v16h bcur[4], bnxt[4];
#pragma unroll
    for (int t = 0; t < 4; ++t) bcur[t] = loadB(0, 0, t);

    for (int chunk = 0; chunk < 8; ++chunk) {
        const int kc = chunk * 128;
        if (chunk) __syncthreads();      // previous chunk fully consumed
        // hidden layer chunk: relu(ea@W1+b1), 32x128 -> f16 LDS
        for (int i = tid; i < 32 * 128; i += 512) {
            int e = i >> 7, k = i & 127;
            float s = b1[kc + k];
#pragma unroll
            for (int f = 0; f < F_E; ++f) s += sh_ea[e][f] * w1[f * KDIM + kc + k];
            sh_hid[e][k] = (_Float16)(s > 0.f ? s : 0.f);
        }
        __syncthreads();

#pragma unroll
        for (int step = 0; step < 4; ++step) {
            // prefetch next K-step's B fragments (software pipeline)
            int ns = step + 1, nc = chunk;
            if (ns == 4) { ns = 0; ++nc; }
            if (nc < 8) {
#pragma unroll
                for (int t = 0; t < 4; ++t) bnxt[t] = loadB(nc, ns, t);
            }

            const int ks = step * 32;
            v16h af[2];
#pragma unroll
            for (int m = 0; m < 2; ++m) {
                const _Float16* p = &sh_hid[m * 16 + l15][ks + half * 8];
                v8h lo = *(const v8h*)p;
                v8h hi = *(const v8h*)(p + 16);
                af[m] = __builtin_shufflevector(lo, hi, 0,1,2,3,4,5,6,7,8,9,10,11,12,13,14,15);
            }
#pragma unroll
            for (int t = 0; t < 4; ++t)
#pragma unroll
                for (int m = 0; m < 2; ++m)
                    acc[m][t] = __builtin_amdgcn_wmma_f32_16x16x32_f16(
                        false, af[m], false, bcur[t], (short)0, acc[m][t], false, false);
#pragma unroll
            for (int t = 0; t < 4; ++t) bcur[t] = bnxt[t];
        }
    }
    __syncthreads();

    // contraction: msg[e,h] += x[src,f] * D ; wave w's tiles all have f == w
#pragma unroll
    for (int m = 0; m < 2; ++m) {
#pragma unroll
        for (int r = 0; r < 8; ++r) {
            int row = m * 16 + r + half * 8;
            float xa = sh_x[row][w];
#pragma unroll
            for (int hb = 0; hb < 4; ++hb) {
                atomicAdd(&sh_msg[row * HDIM + hb * 16 + l15], acc[m][hb][r] * xa); // ds_add_f32
            }
        }
    }
    __syncthreads();

    // add b2 contribution and scatter to agg[dst]
    for (int i = tid; i < 32 * HDIM; i += 512) {
        int e = i >> 6, h = i & 63;
        long eg = eb + e;
        if (eg >= Ee) continue;
        float v = sh_msg[i];
#pragma unroll
        for (int f = 0; f < F_IN; ++f) v += sh_x[e][f] * b2[f * HDIM + h];
        int d = dstI[eg];
        atomicAdd(&agg[(long)d * HDIM + h], v);
    }
    if (tid < 32) {
        long eg = eb + tid;
        if (eg < Ee) atomicAdd(&deg[dstI[eg]], 1.0f);
    }
}

// ---------------- node update: h1 = relu(x@root_w + agg/deg + b); stats for LN1
__global__ __launch_bounds__(256) void k_node1(
    const float* __restrict__ x, const float* __restrict__ rootw,
    const float* __restrict__ cb, const float* __restrict__ agg,
    const float* __restrict__ deg, float* __restrict__ h1,
    float* __restrict__ stats, int Nn) {
    __shared__ float bs[2];
    int tid = threadIdx.x;
    if (tid < 2) bs[tid] = 0.f;
    __syncthreads();
    long i = (long)blockIdx.x * 256 + tid;
    float val = 0.f;
    if (i < (long)Nn * HDIM) {
        int n = (int)(i >> 6), h = (int)(i & 63);
        float s = cb[h];
#pragma unroll
        for (int f = 0; f < F_IN; ++f) s += x[(long)n * F_IN + f] * rootw[f * HDIM + h];
        float dg = deg[n]; dg = dg > 1.f ? dg : 1.f;
        s += agg[i] / dg;
        val = s > 0.f ? s : 0.f;
        h1[i] = val;
    }
    atomicAdd(&bs[0], val);
    atomicAdd(&bs[1], val * val);
    __syncthreads();
    if (tid == 0) { atomicAdd(&stats[0], bs[0]); atomicAdd(&stats[1], bs[1]); }
}

__global__ void k_finstats(float* stats, int base, float cnt) {
    float mu = stats[base] / cnt;
    float var = stats[base + 1] / cnt - mu * mu;
    var = var > 0.f ? var : 0.f;
    stats[base + 2] = mu;
    stats[base + 3] = 1.f / (sqrtf(var) + EPSLN);
}

// ---------------- GAT linear: xh = LN1(h1) @ gat_w  (WMMA)
__global__ __launch_bounds__(256) void k_gatlin(
    const float* __restrict__ h1, const _Float16* __restrict__ gwt,
    const float* __restrict__ nw, const float* __restrict__ nb,
    const float* __restrict__ stats, float* __restrict__ xh, int Nn) {
    int lane = threadIdx.x & 31, w = threadIdx.x >> 5;
    int ntiles = (Nn + 15) >> 4;
    int tile = blockIdx.x * 8 + w;
    if (tile >= ntiles) return;   // wave-uniform exit
    float mu = stats[2], inv = stats[3];
    int half = lane >> 4, l15 = lane & 15;
    int r = tile * 16 + l15;

    v8f acc[4];
#pragma unroll
    for (int t = 0; t < 4; ++t) acc[t] = v8f_zero();

#pragma unroll
    for (int step = 0; step < 2; ++step) {
        v16h a;
#pragma unroll
        for (int j = 0; j < 16; ++j) {
            int k = step * 32 + half * 8 + (j & 7) + ((j >> 3) * 16);
            float v = 0.f;
            if (r < Nn) v = (h1[(long)r * HDIM + k] - mu) * inv * nw[k] + nb[k];
            a[j] = (_Float16)v;
        }
#pragma unroll
        for (int t = 0; t < 4; ++t) {
            const _Float16* bp = gwt + (t * 16 + l15) * HDIM + step * 32 + half * 8;
            v8h blo = *(const v8h*)bp;
            v8h bhi = *(const v8h*)(bp + 16);
            v16h b = __builtin_shufflevector(blo, bhi, 0,1,2,3,4,5,6,7,8,9,10,11,12,13,14,15);
            acc[t] = __builtin_amdgcn_wmma_f32_16x16x32_f16(
                false, a, false, b, (short)0, acc[t], false, false);
        }
    }
#pragma unroll
    for (int t = 0; t < 4; ++t)
#pragma unroll
        for (int rr = 0; rr < 8; ++rr) {
            int row = tile * 16 + rr + half * 8;
            if (row < Nn) xh[(long)row * HDIM + t * 16 + l15] = acc[t][rr];
        }
}

// ---------------- attention vectors
__global__ void k_attvec(const float* xh, const float* aws, const float* awd,
                         float* a_src, float* a_dst, int Nn) {
    long i = (long)blockIdx.x * 256 + threadIdx.x;
    if (i >= (long)Nn * HEADS) return;
    int n = (int)(i >> 2), hd = (int)(i & 3);
    float s1 = 0.f, s2 = 0.f;
#pragma unroll
    for (int d = 0; d < DHEAD; ++d) {
        float v = xh[(long)n * HDIM + hd * 16 + d];
        s1 += v * aws[hd * 16 + d];
        s2 += v * awd[hd * 16 + d];
    }
    a_src[i] = s1; a_dst[i] = s2;
}

__global__ void k_attmax(const int* ei, const float* a_src, const float* a_dst,
                         float* amax, int Nn, int Ee) {
    long i = (long)blockIdx.x * 256 + threadIdx.x;
    if (i >= (long)Ee + Nn) return;
    int s, d;
    if (i < Ee) { s = ei[i]; d = ei[(long)Ee + i]; } else { s = d = (int)(i - Ee); }
#pragma unroll
    for (int hd = 0; hd < HEADS; ++hd) {
        float al = a_src[(long)s * 4 + hd] + a_dst[(long)d * 4 + hd];
        al = al > 0.f ? al : 0.2f * al;
        atomicMaxFloat(&amax[(long)d * 4 + hd], al);
    }
}

__global__ void k_attsum(const int* ei, const float* a_src, const float* a_dst,
                         const float* amax, const float* xh,
                         float* asum, float* attout, int Nn, int Ee) {
    long i = (long)blockIdx.x * 256 + threadIdx.x;
    if (i >= (long)Ee + Nn) return;
    int s, d;
    if (i < Ee) { s = ei[i]; d = ei[(long)Ee + i]; } else { s = d = (int)(i - Ee); }
#pragma unroll
    for (int hd = 0; hd < HEADS; ++hd) {
        float al = a_src[(long)s * 4 + hd] + a_dst[(long)d * 4 + hd];
        al = al > 0.f ? al : 0.2f * al;
        float e = __expf(al - amax[(long)d * 4 + hd]);
        atomicAdd(&asum[(long)d * 4 + hd], e);
#pragma unroll
        for (int k = 0; k < DHEAD; ++k)
            atomicAdd(&attout[(long)d * HDIM + hd * 16 + k],
                      e * xh[(long)s * HDIM + hd * 16 + k]);
    }
}

// ---------------- node2: h2 = relu(attout/asum + gat_b) in-place; stats for LN2
__global__ __launch_bounds__(256) void k_node2(
    float* __restrict__ attout, const float* __restrict__ asum,
    const float* __restrict__ gb, float* __restrict__ stats, int Nn) {
    __shared__ float bs[2];
    int tid = threadIdx.x;
    if (tid < 2) bs[tid] = 0.f;
    __syncthreads();
    long i = (long)blockIdx.x * 256 + tid;
    float val = 0.f;
    if (i < (long)Nn * HDIM) {
        int n = (int)(i >> 6), c = (int)(i & 63), hd = c >> 4;
        float v = attout[i] / (asum[(long)n * 4 + hd] + 1e-16f) + gb[c];
        val = v > 0.f ? v : 0.f;
        attout[i] = val;
    }
    atomicAdd(&bs[0], val);
    atomicAdd(&bs[1], val * val);
    __syncthreads();
    if (tid == 0) { atomicAdd(&stats[4], bs[0]); atomicAdd(&stats[5], bs[1]); }
}

// ---------------- pooling: LN2(h2) segment-sum by batch
__global__ void k_pool(const float* h2, const float* nw, const float* nb,
                       const float* stats, const int* batch,
                       float* gs, float* cnt, int Nn) {
    long i = (long)blockIdx.x * 256 + threadIdx.x;
    if (i >= (long)Nn * HDIM) return;
    int n = (int)(i >> 6), c = (int)(i & 63);
    float v = (h2[i] - stats[6]) * stats[7] * nw[c] + nb[c];
    int g = batch[n];
    atomicAdd(&gs[g * HDIM + c], v);
    if (c == 0) atomicAdd(&cnt[g], 1.f);
}

__global__ void k_head(const float* gs, const float* cnt, const float* lw,
                       const float* lb, float* out) {
    int g = threadIdx.x;
    if (g >= NGRP) return;
    float cn = cnt[g]; cn = cn > 1.f ? cn : 1.f;
    float s = 0.f;
#pragma unroll
    for (int c = 0; c < HDIM; ++c) s += (gs[g * HDIM + c] / cn) * lw[c];
    out[g] = s + lb[0];
}

extern "C" void kernel_launch(void* const* d_in, const int* in_sizes, int n_in,
                              void* d_out, int out_size, void* d_ws, size_t ws_size,
                              hipStream_t stream) {
    const float* x     = (const float*)d_in[0];
    const int*   ei    = (const int*)d_in[1];
    const float* ea    = (const float*)d_in[2];
    const int*   batch = (const int*)d_in[3];
    const float* w1    = (const float*)d_in[4];
    const float* b1    = (const float*)d_in[5];
    const float* w2    = (const float*)d_in[6];
    const float* b2    = (const float*)d_in[7];
    const float* rootw = (const float*)d_in[8];
    const float* cb    = (const float*)d_in[9];
    const float* n1w   = (const float*)d_in[10];
    const float* n1b   = (const float*)d_in[11];
    const float* gatw  = (const float*)d_in[12];
    const float* aws   = (const float*)d_in[13];
    const float* awd   = (const float*)d_in[14];
    const float* gb    = (const float*)d_in[15];
    const float* n2w   = (const float*)d_in[16];
    const float* n2b   = (const float*)d_in[17];
    const float* lw    = (const float*)d_in[18];
    const float* lb    = (const float*)d_in[19];
    float* out = (float*)d_out;

    const int Nn = in_sizes[0] / F_IN;
    const int Ee = in_sizes[1] / 2;

    // ----- workspace layout (zero-init region first, contiguous)
    char* ws = (char*)d_ws;
    size_t off = 0;
    auto alloc = [&](size_t bytes) -> char* {
        char* p = ws + off;
        off = (off + bytes + 255) & ~(size_t)255;
        return p;
    };
    float* agg    = (float*)alloc((size_t)Nn * HDIM * 4);
    float* deg    = (float*)alloc((size_t)Nn * 4);
    float* asum   = (float*)alloc((size_t)Nn * HEADS * 4);
    float* attout = (float*)alloc((size_t)Nn * HDIM * 4);
    float* gs     = (float*)alloc((size_t)NGRP * HDIM * 4);
    float* cnt    = (float*)alloc((size_t)NGRP * 4);
    float* stats  = (float*)alloc(8 * 4);
    size_t zbytes = off;                       // everything above gets zeroed
    float* amax   = (float*)alloc((size_t)Nn * HEADS * 4);
    float* h1     = (float*)alloc((size_t)Nn * HDIM * 4);
    float* xh     = (float*)alloc((size_t)Nn * HDIM * 4);
    float* a_src  = (float*)alloc((size_t)Nn * HEADS * 4);
    float* a_dst  = (float*)alloc((size_t)Nn * HEADS * 4);
    _Float16* w2t = (_Float16*)alloc((size_t)KDIM * KDIM * 2);
    _Float16* gwt = (_Float16*)alloc((size_t)HDIM * HDIM * 2);

    // ----- launches
    {
        int tot = KDIM * KDIM + HDIM * HDIM;
        k_prep<<<(tot + 255) / 256, 256, 0, stream>>>(w2, gatw, w2t, gwt);
    }
    k_zero<<<2048, 256, 0, stream>>>((float*)ws, (long)(zbytes / 4));
    {
        long na = (long)Nn * HEADS;
        k_fill<<<(int)((na + 255) / 256), 256, 0, stream>>>(amax, na, -1e30f);
    }
    k_nnconv<<<(Ee + 31) / 32, 512, 0, stream>>>(x, ei, ea, w1, b1, b2, w2t, agg, deg, Nn, Ee);
    {
        long ne = (long)Nn * HDIM;
        k_node1<<<(int)((ne + 255) / 256), 256, 0, stream>>>(x, rootw, cb, agg, deg, h1, stats, Nn);
    }
    k_finstats<<<1, 1, 0, stream>>>(stats, 0, (float)((long)Nn * HDIM));
    {
        int ntiles = (Nn + 15) / 16;
        k_gatlin<<<(ntiles + 7) / 8, 256, 0, stream>>>(h1, gwt, n1w, n1b, stats, xh, Nn);
    }
    {
        long na = (long)Nn * HEADS;
        k_attvec<<<(int)((na + 255) / 256), 256, 0, stream>>>(xh, aws, awd, a_src, a_dst, Nn);
    }
    {
        long nt = (long)Ee + Nn;
        k_attmax<<<(int)((nt + 255) / 256), 256, 0, stream>>>(ei, a_src, a_dst, amax, Nn, Ee);
        k_attsum<<<(int)((nt + 255) / 256), 256, 0, stream>>>(ei, a_src, a_dst, amax, xh, asum, attout, Nn, Ee);
    }
    {
        long ne = (long)Nn * HDIM;
        k_node2<<<(int)((ne + 255) / 256), 256, 0, stream>>>(attout, asum, gb, stats, Nn);
    }
    k_finstats<<<1, 1, 0, stream>>>(stats, 4, (float)((long)Nn * HDIM));
    {
        long ne = (long)Nn * HDIM;
        k_pool<<<(int)((ne + 255) / 256), 256, 0, stream>>>(attout, n2w, n2b, stats, batch, gs, cnt, Nn);
    }
    k_head<<<1, 64, 0, stream>>>(gs, cnt, lw, lb, out);
}
